// ScaledDotProductAttention2_71528385347805
// MI455X (gfx1250) — matrix operand order repaired and checked by reading, not compiled, not run on hardware
//
#include <hip/hip_runtime.h>
#include <stdint.h>

// ---- CDNA5 WMMA vector types (wave32) ----
typedef __attribute__((ext_vector_type(16))) __bf16 v16bf;
typedef __attribute__((ext_vector_type(8)))  float  v8f;

#define BH       96
#define L        1024
#define D        64
#define NTILES   (L / 16)        // 64 k-tiles in the QK^T pass
#define STRIDE   1028            // padded LDS row stride (floats): 1028%64=4 -> conflict-free
#define NEG_INF  (-1.0e10f)
#define INV_TEMP 0.125f          // 1 / 8.0

union BF8x2 { unsigned u[8]; v16bf v; };

// two fp32 -> packed bf16 (truncate) in ONE v_perm_b32
static __device__ __forceinline__ unsigned pack_bf16x2(float lo, float hi) {
  // {src0,src1} = {hi,lo}; pick bytes {7,6,3,2} -> [31:16]=hi.bf16, [15:0]=lo.bf16
  return __builtin_amdgcn_perm(__float_as_uint(hi), __float_as_uint(lo), 0x07060302u);
}

// A-matrix (16x32 bf16) K-offset for pair-register p, half h (contiguous k = 16h + 2p, matches B)
static __device__ __forceinline__ int a_klocal(int p, int h) {
  return 16 * h + 2 * p;
}

__global__ void __launch_bounds__(32)
fused_attention_kernel(const float* __restrict__ Q,
                       const float* __restrict__ K,
                       const float* __restrict__ V,
                       const unsigned char* __restrict__ mask,
                       float* __restrict__ out,     // [BH, L, D]
                       float* __restrict__ attn)    // [BH, L, L]
{
  extern __shared__ float smem_f[];
  float* srow = smem_f;                 // 16 x STRIDE: raw scores, then exp(s-max)
  float* smax = smem_f + 16 * STRIDE;   // 16 row maxima
  float* sinv = smax + 16;              // 16 reciprocal row sums

  const int lane = threadIdx.x;         // 0..31 (wave32)
  const int Ml   = lane & 15;
  const int h    = lane >> 4;

  const int tile = blockIdx.x;          // 0 .. BH*(L/16)-1
  const int b    = tile >> 6;           // tile / 64
  const int q0   = (tile & 63) << 4;

  // ---------- load Q tile as two 16x32 bf16 A-fragments ----------
  BF8x2 aQ[2];
  {
    const size_t qbase = ((size_t)b * L + q0 + Ml) * D;
#pragma unroll
    for (int f = 0; f < 2; ++f) {
#pragma unroll
      for (int p = 0; p < 8; ++p) {
        const int kl = a_klocal(p, h);
        const float2 t = *(const float2*)(Q + qbase + 32 * f + kl);
        aQ[f].u[p] = pack_bf16x2(t.x, t.y);
      }
    }
  }

  float m[8];
#pragma unroll
  for (int r = 0; r < 8; ++r) m[r] = -3.0e38f;

  // ---------- phase 1: S = mask ? -1e10 : (Q K^T)/8 ; stage in LDS; max only ----------
  for (int nt = 0; nt < NTILES; ++nt) {
    const int n0 = nt * 16;
    BF8x2 bK[2];
    const size_t kbase = ((size_t)b * L + n0 + Ml) * D;   // B col N = Ml
    if (nt + 1 < NTILES)
      __builtin_prefetch(K + kbase + (size_t)16 * D, 0, 0);   // global_prefetch_b8
#pragma unroll
    for (int f = 0; f < 2; ++f) {
#pragma unroll
      for (int vv = 0; vv < 8; ++vv) {
        const int dl = 16 * h + 2 * vv;                   // B 32x16 layout
        const float2 t = *(const float2*)(K + kbase + 32 * f + dl);
        bK[f].u[vv] = pack_bf16x2(t.x, t.y);
      }
    }
    v8f c = {0.f, 0.f, 0.f, 0.f, 0.f, 0.f, 0.f, 0.f};
    c = __builtin_amdgcn_wmma_f32_16x16x32_bf16(false, aQ[0].v, false, bK[0].v,
                                                (short)0, c, false, false);
    c = __builtin_amdgcn_wmma_f32_16x16x32_bf16(false, aQ[1].v, false, bK[1].v,
                                                (short)0, c, false, false);
    const int col = n0 + Ml;
    const size_t mbase = ((size_t)b * L + q0) * L + col;
#pragma unroll
    for (int r = 0; r < 8; ++r) {
      const int row = r + 8 * h;                          // C layout
      float s = c[r] * INV_TEMP;
      if (mask[mbase + (size_t)row * L]) s = NEG_INF;
      srow[row * STRIDE + col] = s;
      m[r] = fmaxf(m[r], s);                              // max only, no exp here
    }
  }

  // ---------- phase 2a: cross-lane row max (16 lanes per half) ----------
#pragma unroll
  for (int r = 0; r < 8; ++r) {
    float M = m[r];
    M = fmaxf(M, __shfl_xor(M, 1, 32));
    M = fmaxf(M, __shfl_xor(M, 2, 32));
    M = fmaxf(M, __shfl_xor(M, 4, 32));
    M = fmaxf(M, __shfl_xor(M, 8, 32));
    if (Ml == 0) smax[r + 8 * h] = M;
  }
  __syncthreads();   // single-wave workgroup: cheap

  // ---------- phase 2b: exponentiate every score exactly once (float4) ----------
  // lane mapping: idx = i*32+lane -> row = idx>>3 (fixed per (i,lane)), 4 cols per step
  {
    float acc[4] = {0.f, 0.f, 0.f, 0.f};
    for (int kt = 0; kt < 32; ++kt) {
      const int k0 = kt * 32;
#pragma unroll
      for (int i = 0; i < 4; ++i) {
        const int idx = i * 32 + lane;
        const int row = idx >> 3;
        const int c0  = k0 + (idx & 7) * 4;
        float4 t = *(const float4*)(srow + row * STRIDE + c0);
        const float mx = smax[row];
        t.x = __expf(t.x - mx); t.y = __expf(t.y - mx);
        t.z = __expf(t.z - mx); t.w = __expf(t.w - mx);
        *(float4*)(srow + row * STRIDE + c0) = t;
        acc[i] += (t.x + t.y) + (t.z + t.w);
      }
    }
#pragma unroll
    for (int i = 0; i < 4; ++i) {        // rows i*4 .. i*4+3, 8 lanes per row
      float S = acc[i];
      S += __shfl_xor(S, 1, 32);
      S += __shfl_xor(S, 2, 32);
      S += __shfl_xor(S, 4, 32);
      if ((lane & 7) == 0) sinv[i * 4 + (lane >> 3)] = 1.0f / S;
    }
  }
  __syncthreads();

  const float myInv = sinv[Ml];

  // ---------- phase 3: coalesced normalized-attn writeback + P @ V ----------
  v8f acc[4];
#pragma unroll
  for (int db = 0; db < 4; ++db)
    acc[db] = (v8f){0.f, 0.f, 0.f, 0.f, 0.f, 0.f, 0.f, 0.f};

  const size_t vrow0 = (size_t)b * L;       // V row base
  for (int kt = 0; kt < 32; ++kt) {
    const int k0 = kt * 32;
    if (kt + 1 < 32)
      __builtin_prefetch(V + (vrow0 + k0 + 32 + lane) * D, 0, 0);

    // normalized attn store: float4 per lane, rows of 128B fully coalesced
    {
      const size_t abase = ((size_t)b * L + q0) * L;
#pragma unroll
      for (int i = 0; i < 4; ++i) {
        const int idx = i * 32 + lane;
        const int row = idx >> 3;
        const int c0  = k0 + (idx & 7) * 4;
        float4 t = *(const float4*)(srow + row * STRIDE + c0);
        const float sc = sinv[row];
        t.x *= sc; t.y *= sc; t.z *= sc; t.w *= sc;
        *(float4*)(attn + abase + (size_t)row * L + c0) = t;
      }
    }

    // A fragment: P tile 16x32 (bf16) = e * (1/sum), e straight from LDS
    BF8x2 aP;
#pragma unroll
    for (int p = 0; p < 8; ++p) {
      const int kl = a_klocal(p, h);
      const float2 t = *(const float2*)(srow + Ml * STRIDE + k0 + kl);
      aP.u[p] = pack_bf16x2(t.x * myInv, t.y * myInv);
    }

    // B fragments: V 32x64 as four 32x16 tiles; 4 WMMAs accumulate 16x64 out
#pragma unroll
    for (int db = 0; db < 4; ++db) {
      BF8x2 bV;
      const int col = db * 16 + Ml;
#pragma unroll
      for (int vv = 0; vv < 8; ++vv) {
        const int kr = k0 + 16 * h + 2 * vv;
        const float x0 = V[(vrow0 + kr)     * D + col];
        const float x1 = V[(vrow0 + kr + 1) * D + col];
        bV.u[vv] = pack_bf16x2(x0, x1);
      }
      acc[db] = __builtin_amdgcn_wmma_f32_16x16x32_bf16(false, aP.v, false, bV.v,
                                                        (short)0, acc[db], false, false);
    }
  }

  // ---------- write 16x64 output tile ----------
#pragma unroll
  for (int db = 0; db < 4; ++db) {
    const int col = db * 16 + Ml;
#pragma unroll
    for (int r = 0; r < 8; ++r) {
      const int row = q0 + r + 8 * h;
      out[((size_t)b * L + row) * D + col] = acc[db][r];
    }
  }
}

extern "C" void kernel_launch(void* const* d_in, const int* in_sizes, int n_in,
                              void* d_out, int out_size, void* d_ws, size_t ws_size,
                              hipStream_t stream) {
  const float* q = (const float*)d_in[0];
  const float* k = (const float*)d_in[1];
  const float* v = (const float*)d_in[2];
  // d_in[3] = bias (unused by reference forward)
  const unsigned char* mask = (const unsigned char*)d_in[4];  // jnp.bool_ -> 1 byte

  float* out  = (float*)d_out;                       // [96,1024,64]
  float* attn = out + (size_t)BH * L * D;            // [96,1024,1024]

  const int smem = (16 * STRIDE + 32) * (int)sizeof(float);   // ~64.4 KB < 320 KB WGP LDS
  (void)hipFuncSetAttribute((const void*)fused_attention_kernel,
                            hipFuncAttributeMaxDynamicSharedMemorySize, smem);

  fused_attention_kernel<<<BH * (L / 16), 32, smem, stream>>>(q, k, v, mask, out, attn);
  (void)in_sizes; (void)n_in; (void)out_size; (void)d_ws; (void)ws_size;
}
